// GRNNTransformGated_55259049230550
// MI455X (gfx1250) — matrix operand order repaired and checked by reading, not compiled, not run on hardware
//
#include <hip/hip_runtime.h>
#include <hip/hip_bf16.h>

typedef __attribute__((ext_vector_type(16))) __bf16 v16bf;
typedef __attribute__((ext_vector_type(8)))  float  v8f;

// ---------- helpers ----------
__device__ __forceinline__ unsigned short f2bf(float x) {
    union { float f; unsigned u; } v; v.f = x;
    unsigned r = v.u + 0x7FFFu + ((v.u >> 16) & 1u);   // round-to-nearest-even
    return (unsigned short)(r >> 16);
}
__device__ __forceinline__ float bf2f(unsigned short b) {
    union { unsigned u; float f; } v; v.u = ((unsigned)b) << 16;
    return v.f;
}

union AF { unsigned u[8]; v16bf v; };
union CF { v8f v; float f[8]; };

__device__ __forceinline__ v8f wmma_bf16(v16bf a, v16bf b, v8f c) {
    return __builtin_amdgcn_wmma_f32_16x16x32_bf16(false, a, false, b, (short)0, c,
                                                   false, false);
}

// A fragment (16x32 bf16, MxK), row-major source split into two column
// segments: cols [0,K0) from A0 (row stride lda0), cols [K0,K) from A1.
// ISA layout: lane m=lane&15, half h=lane>>4; pair p<4 -> K = 8h+2p,
// p>=4 -> K = 16+8h+2(p-4). Pairs are K-consecutive -> dword loads that
// the compiler merges into two b128 loads per fragment.
__device__ __forceinline__ v16bf wmma_load_a2(const unsigned short* A0, int lda0, int K0,
                                              const unsigned short* A1, int lda1,
                                              int row, int kbase, int lane) {
    int m = lane & 15, h = lane >> 4;
    const unsigned short* r0 = A0 + (size_t)(row + m) * lda0;
    const unsigned short* r1 = A1 + (size_t)(row + m) * lda1;
    AF fa;
#pragma unroll
    for (int p = 0; p < 8; ++p) {
        int k0 = ((p < 4) ? (2 * p) : (16 + 2 * (p - 4))) + 8 * h;
        int k  = kbase + k0;
        const unsigned short* s = (k < K0) ? (r0 + k) : (r1 + (k - K0));
        fa.u[p] = *(const unsigned*)s;            // 4B aligned: k even, even strides
    }
    return fa.v;
}

// B fragment (32x16 bf16, KxN) from pair-packed weights:
// Bp[k2*N + n] = {bf16 W[2k2,n], bf16 W[2k2+1,n]}.
// ISA layout: lane n=lane&15, half h: VGPR p<4 holds K = 8h+2p,+1; p>=4 holds K = 16+8h+2(p-4),+1.
__device__ __forceinline__ v16bf wmma_load_b(const unsigned* Bp, int N, int col,
                                             int kbase, int lane) {
    int n = lane & 15, h = lane >> 4;
    const unsigned* base = Bp + (size_t)((kbase >> 1) + 4 * h) * N + (col + n);
    AF fb;
#pragma unroll
    for (int p = 0; p < 8; ++p) fb.u[p] = base[(size_t)((p < 4) ? p : (p + 4)) * N];
    return fb.v;
}

// ---------- weight conversion / K-pair packing ----------
__global__ __launch_bounds__(256)
void pack_weights(const float* __restrict__ Wr, const float* __restrict__ Wh,
                  const float* __restrict__ Wz,
                  unsigned* __restrict__ Wrp, unsigned* __restrict__ Whp,
                  unsigned* __restrict__ Wzp) {
    int t = blockIdx.x * 256 + threadIdx.x;
    const float* W; unsigned* P; int N; int id;
    if (t < 73728)               { W = Wr; P = Wrp; N = 384; id = t; }
    else if (t < 73728 + 24576)  { W = Wh; P = Whp; N = 128; id = t - 73728; }
    else if (t < 229376)         { W = Wz; P = Wzp; N = 512; id = t - 98304; }
    else return;
    int k2 = id / N, n = id % N;
    unsigned lo = f2bf(W[(size_t)(2 * k2) * N + n]);
    unsigned hi = f2bf(W[(size_t)(2 * k2 + 1) * N + n]);
    P[(size_t)k2 * N + n] = lo | (hi << 16);
}

// ---------- per-level: u = conv_chain(c@W_u+b_u); build hhu ----------
__global__ __launch_bounds__(256)
void u_hhu(const float* __restrict__ contents,      // n x 7 (level slice)
           const float* __restrict__ W_u, const float* __restrict__ b_u,
           const float* __restrict__ conv_w, const float* __restrict__ conv_b,
           const float* __restrict__ emb_prev,      // child embeddings (f32)
           const int* __restrict__ children,        // pairs for this level
           unsigned short* __restrict__ hhu,        // n x 384 bf16 (internal)
           float* __restrict__ emb_out,             // leaf: n x 128 f32
           int n, int leaf) {
    int t = blockIdx.x * 256 + threadIdx.x;
    if (t >= n * 128) return;
    int node = t >> 7, f = t & 127;
    const float* c = contents + (size_t)node * 7;
    float v = b_u[f];
#pragma unroll
    for (int k = 0; k < 7; ++k) v = fmaf(c[k], W_u[k * 128 + f], v);
    float cw = conv_w[0], cb = conv_b[0];
#pragma unroll
    for (int q = 0; q < 3; ++q) { v = cw * v + cb; v = v > 0.f ? v : 0.f; }
    if (leaf) {
        emb_out[(size_t)node * 128 + f] = v;
    } else {
        int cL = children[2 * node], cR = children[2 * node + 1];
        size_t r = (size_t)node * 384;
        hhu[r + f]       = f2bf(emb_prev[(size_t)cL * 128 + f]);
        hhu[r + 128 + f] = f2bf(emb_prev[(size_t)cR * 128 + f]);
        hhu[r + 256 + f] = f2bf(v);
    }
}

// ---------- GEMM + epilogue (MODE 0: sigmoid*mul -> rh; MODE 1: conv_chain -> hH)
// Waves arranged 4(M) x 2(N) per 256-thread block; each wave: 2 M-tiles x 4 N-tiles.
// Block footprint: 128 rows x 128 cols. N must be a multiple of 128.
template <int MODE>
__global__ __launch_bounds__(256)
void gemm_epi(const unsigned short* __restrict__ A, int K,       // n x K bf16
              const unsigned* __restrict__ Bp, int N,            // packed K/2 x N
              const float* __restrict__ bias,
              const unsigned short* __restrict__ Emul,           // MODE0: hhu (n x N)
              const float* __restrict__ conv_w, const float* __restrict__ conv_b,
              unsigned short* __restrict__ out) {
    int lane  = threadIdx.x & 31;
    int wave  = threadIdx.x >> 5;
    int waveM = wave & 3, waveN = wave >> 2;
    int rowBase = blockIdx.x * 128 + waveM * 32;
    int colBase = blockIdx.y * 128 + waveN * 64;   // 4 N-tiles: +0,16,32,48
    v8f acc[2][4] = {};
    for (int kb = 0; kb < K; kb += 32) {
        v16bf a0 = wmma_load_a2(A, K, K, A, K, rowBase,      kb, lane);
        v16bf a1 = wmma_load_a2(A, K, K, A, K, rowBase + 16, kb, lane);
#pragma unroll
        for (int g = 0; g < 4; ++g) {
            v16bf b = wmma_load_b(Bp, N, colBase + g * 16, kb, lane);
            acc[0][g] = wmma_bf16(a0, b, acc[0][g]);
            acc[1][g] = wmma_bf16(a1, b, acc[1][g]);
        }
    }
    int h = lane >> 4;
    float cw = 0.f, cb = 0.f;
    if (MODE == 1) { cw = conv_w[0]; cb = conv_b[0]; }
#pragma unroll
    for (int g = 0; g < 4; ++g) {
        int col = colBase + g * 16 + (lane & 15);
        float bs = bias[col];
#pragma unroll
        for (int t2 = 0; t2 < 2; ++t2) {
            CF cf; cf.v = acc[t2][g];
#pragma unroll
            for (int i = 0; i < 8; ++i) {
                int m = rowBase + t2 * 16 + i + 8 * h;   // C/D layout: VGPR i -> M=i+8h
                float v = cf.f[i] + bs;
                if (MODE == 0) {
                    v = (1.f / (1.f + __expf(-v))) * bf2f(Emul[(size_t)m * N + col]);
                } else {
#pragma unroll
                    for (int q = 0; q < 3; ++q) { v = cw * v + cb; v = v > 0.f ? v : 0.f; }
                }
                out[(size_t)m * N + col] = f2bf(v);
            }
        }
    }
}

// ---------- z GEMM (K=512 split A) + 4-way softmax gate + combine
// Waves 4(M) x 2(F); each wave: 2 M-tiles x 4 gate tiles (cols f,f+128,f+256,f+384).
// Block footprint: 128 rows x 32 features. grid.y = 128/32 = 4.
__global__ __launch_bounds__(256)
void gemm_gate(const unsigned short* __restrict__ hHb,   // n x 128 bf16
               const unsigned short* __restrict__ hhu,   // n x 384 bf16
               const unsigned* __restrict__ Bp,          // Wz packed 256 x 512
               const float* __restrict__ bias,           // 512
               const float* __restrict__ emb_prev,       // child level f32
               const int* __restrict__ children,
               float* __restrict__ out) {                // n x 128 f32
    int lane  = threadIdx.x & 31;
    int wave  = threadIdx.x >> 5;
    int waveM = wave & 3, waveF = wave >> 2;
    int rowBase = blockIdx.x * 128 + waveM * 32;
    int colBase = blockIdx.y * 32 + waveF * 16;          // feature tile, 0..112
    v8f acc[2][4] = {};
    for (int kb = 0; kb < 512; kb += 32) {
        v16bf a0 = wmma_load_a2(hHb, 128, 128, hhu, 384, rowBase,      kb, lane);
        v16bf a1 = wmma_load_a2(hHb, 128, 128, hhu, 384, rowBase + 16, kb, lane);
#pragma unroll
        for (int g = 0; g < 4; ++g) {                    // gate tiles f + 128g
            v16bf b = wmma_load_b(Bp, 512, colBase + g * 128, kb, lane);
            acc[0][g] = wmma_bf16(a0, b, acc[0][g]);
            acc[1][g] = wmma_bf16(a1, b, acc[1][g]);
        }
    }
    int h = lane >> 4;
    int f = colBase + (lane & 15);
    float bz0 = bias[f], bz1 = bias[f + 128], bz2 = bias[f + 256], bz3 = bias[f + 384];
#pragma unroll
    for (int t2 = 0; t2 < 2; ++t2) {
        CF z0, z1, z2, z3;
        z0.v = acc[t2][0]; z1.v = acc[t2][1]; z2.v = acc[t2][2]; z3.v = acc[t2][3];
#pragma unroll
        for (int i = 0; i < 8; ++i) {
            int m = rowBase + t2 * 16 + i + 8 * h;
            float a = z0.f[i] + bz0, b = z1.f[i] + bz1;
            float c = z2.f[i] + bz2, d = z3.f[i] + bz3;
            float mx = fmaxf(fmaxf(a, b), fmaxf(c, d));
            float e0 = __expf(a - mx), e1 = __expf(b - mx);
            float e2 = __expf(c - mx), e3 = __expf(d - mx);
            float inv = 1.f / (e0 + e1 + e2 + e3);
            int cL = children[2 * m], cR = children[2 * m + 1];
            float hH = bf2f(hHb[(size_t)m * 128 + f]);
            float hL = emb_prev[(size_t)cL * 128 + f];
            float hR = emb_prev[(size_t)cR * 128 + f];
            float uu = bf2f(hhu[(size_t)m * 384 + 256 + f]);
            out[(size_t)m * 128 + f] = (e0 * hH + e1 * hL + e2 * hR + e3 * uu) * inv;
        }
    }
}

// ---------- host orchestration ----------
extern "C" void kernel_launch(void* const* d_in, const int* in_sizes, int n_in,
                              void* d_out, int out_size, void* d_ws, size_t ws_size,
                              hipStream_t stream) {
    const float* contents = (const float*)d_in[0];
    const int*   children = (const int*)d_in[1];
    const float* W_u = (const float*)d_in[2];
    const float* b_u = (const float*)d_in[3];
    const float* W_h = (const float*)d_in[4];
    const float* b_h = (const float*)d_in[5];
    const float* W_z = (const float*)d_in[6];
    const float* b_z = (const float*)d_in[7];
    const float* W_r = (const float*)d_in[8];
    const float* b_r = (const float*)d_in[9];
    const float* conv_w = (const float*)d_in[10];
    const float* conv_b = (const float*)d_in[11];

    char* wsp = (char*)d_ws;
    auto carve = [&](size_t bytes) -> char* {
        char* p = wsp; wsp += (bytes + 255) & ~(size_t)255; return p;
    };
    unsigned* Wrp = (unsigned*)carve(294912);                      // 192x384 u32
    unsigned* Whp = (unsigned*)carve(98304);                       // 192x128 u32
    unsigned* Wzp = (unsigned*)carve(524288);                      // 256x512 u32
    float* emb0 = (float*)carve((size_t)524288 * 128 * 4);
    float* emb1 = (float*)carve((size_t)262144 * 128 * 4);
    unsigned short* hhu = (unsigned short*)carve((size_t)262144 * 384 * 2);
    unsigned short* rh  = (unsigned short*)carve((size_t)262144 * 384 * 2);
    unsigned short* hHb = (unsigned short*)carve((size_t)262144 * 128 * 2);

    pack_weights<<<(229376 + 255) / 256, 256, 0, stream>>>(W_r, W_h, W_z, Wrp, Whp, Wzp);

    long long off[11], coff[10];
    off[0] = 0;
    for (int j = 0; j < 10; ++j) off[j + 1] = off[j] + (1024LL << j);
    coff[0] = 0;
    for (int j = 0; j < 9; ++j) coff[j + 1] = coff[j] + (1024LL << j);

    float* embPrev = emb1;   // unused at leaf
    float* embCur  = emb0;
    for (int j = 9; j >= 0; --j) {
        long long n = 1024LL << j;
        const float* cslice = contents + off[j] * 7;
        if (j == 9) {
            u_hhu<<<(unsigned)((n * 128 + 255) / 256), 256, 0, stream>>>(
                cslice, W_u, b_u, conv_w, conv_b,
                nullptr, nullptr, nullptr, embCur, (int)n, 1);
        } else {
            const int* ch = children + coff[j] * 2;
            u_hhu<<<(unsigned)((n * 128 + 255) / 256), 256, 0, stream>>>(
                cslice, W_u, b_u, conv_w, conv_b,
                embPrev, ch, hhu, nullptr, (int)n, 0);

            dim3 g1((unsigned)(n / 128), 3);    // r-gate * hhu -> rh  (N=384)
            gemm_epi<0><<<g1, 256, 0, stream>>>(hhu, 384, Wrp, 384, b_r, hhu,
                                                conv_w, conv_b, rh);
            dim3 g2((unsigned)(n / 128), 1);    // conv_chain(rh@W_h) -> hH (N=128)
            gemm_epi<1><<<g2, 256, 0, stream>>>(rh, 384, Whp, 128, b_h, rh,
                                                conv_w, conv_b, hHb);
            float* outp = (j == 0) ? (float*)d_out : embCur;
            dim3 g3((unsigned)(n / 128), 4);    // z GEMM + softmax gate + combine
            gemm_gate<<<g3, 256, 0, stream>>>(hHb, hhu, Wzp, b_z, embPrev, ch, outp);
        }
        float* t = embPrev; embPrev = embCur; embCur = t;
    }
}
